// SelfAttention_87952340288008
// MI455X (gfx1250) — matrix operand order repaired and checked by reading, not compiled, not run on hardware
//
#include <hip/hip_runtime.h>

typedef _Float16 v16h __attribute__((ext_vector_type(16)));
typedef _Float16 v8h  __attribute__((ext_vector_type(8)));
typedef float    v8f  __attribute__((ext_vector_type(8)));

#define S_LEN 2048
#define D_INF 1024

static __device__ __forceinline__ v8f wmma16(v16h a, v16h b, v8f c) {
  // D = A(16x32 f16) * B(32x16 f16) + C(16x16 f32)
  return __builtin_amdgcn_wmma_f32_16x16x32_f16(false, a, false, b, (short)0, c,
                                                false, false);
}

static __device__ __forceinline__ v16h cat16(v8h lo, v8h hi) {
  return __builtin_shufflevector(lo, hi, 0, 1, 2, 3, 4, 5, 6, 7, 8, 9, 10, 11,
                                 12, 13, 14, 15);
}

static __device__ __forceinline__ v8h cvt8(const float* __restrict__ p) {
  v8h r;
#pragma unroll
  for (int i = 0; i < 8; ++i) r[i] = (_Float16)p[i];
  return r;
}

// ---------------------------------------------------------------------------
// Kernel 0: transpose + f16-convert weights. WT[j][k] = W(j)[k][j%64]
// rows 0-63: Wq^T, 64-127: Wk^T, 128-191: Wv^T  (makes B-fragments contiguous)
// ---------------------------------------------------------------------------
__global__ __launch_bounds__(256) void wt_kernel(const float* __restrict__ Wq,
                                                 const float* __restrict__ Wk,
                                                 const float* __restrict__ Wv,
                                                 _Float16* __restrict__ WT) {
  int idx = blockIdx.x * 256 + threadIdx.x;  // 0 .. 192*1024-1
  int j = idx >> 10, k = idx & 1023;
  const float* W = (j < 64) ? Wq : ((j < 128) ? Wk : Wv);
  WT[idx] = (_Float16)W[k * 64 + (j & 63)];
}

// ---------------------------------------------------------------------------
// Kernel 1: fused QKV projection. One wave per 16-row tile, 12 C-fragments
// (4 Q n-tiles, 4 K n-tiles, 4 V n-tiles), K-loop of 32 over D_IN=1024.
// Q scaled by 1/sqrt(64); V stored transposed Vt[b][64][S].
// ---------------------------------------------------------------------------
__global__ __launch_bounds__(256) void qkv_kernel(
    const float* __restrict__ X, const _Float16* __restrict__ WT,
    _Float16* __restrict__ Qb, _Float16* __restrict__ Kb,
    _Float16* __restrict__ Vt) {
  const int lane = threadIdx.x & 31;
  const int w = threadIdx.x >> 5;
  const int g = lane >> 4, ln = lane & 15;
  const int tile = blockIdx.x * 8 + w;  // 0..1023
  const int row0 = tile * 16;

  v8f acc[12];
#pragma unroll
  for (int i = 0; i < 12; ++i)
#pragma unroll
    for (int j = 0; j < 8; ++j) acc[i][j] = 0.0f;

  const float* xrow = X + (size_t)(row0 + ln) * D_INF;  // this lane's M-row

  for (int k0 = 0; k0 < D_INF; k0 += 32) {
    // A-fragment (16-bit A 16x32): lane<16 K=0..15 ; lane>=16 K=16..31
    v16h a = cat16(cvt8(xrow + k0 + 16 * g), cvt8(xrow + k0 + 16 * g + 8));
#pragma unroll
    for (int nt = 0; nt < 12; ++nt) {
      // B-fragment: lane<16 holds K=0..15 of column (16nt+ln), lane>=16 K=16..31
      v16h b = *(const v16h*)(WT + (size_t)(16 * nt + ln) * D_INF + k0 + 16 * g);
      acc[nt] = wmma16(a, b, acc[nt]);
    }
  }

  // Epilogue: C layout element (r,lane): M = r + 8*(lane>>4), N = lane&15
#pragma unroll
  for (int nt = 0; nt < 12; ++nt) {
#pragma unroll
    for (int r = 0; r < 8; ++r) {
      int row = row0 + r + 8 * g;
      float v = acc[nt][r];
      if (nt < 4) {
        Qb[(size_t)row * 64 + nt * 16 + ln] = (_Float16)(v * 0.125f);
      } else if (nt < 8) {
        Kb[(size_t)row * 64 + (nt - 4) * 16 + ln] = (_Float16)v;
      } else {
        int vd = (nt - 8) * 16 + ln;
        int bb = row >> 11;       // row / 2048
        int s = row & 2047;
        Vt[((size_t)bb * 64 + vd) * S_LEN + s] = (_Float16)v;
      }
    }
  }
}

// ---------------------------------------------------------------------------
// Kernel 2: causal flash attention. One wave per 16-query tile; key blocks of
// 32 (2 score WMMAs over 2 K-steps each + 4 PV WMMAs per block). Softmax
// stats live in C-fragment row alignment; only P goes through LDS (C->A).
// ---------------------------------------------------------------------------
__global__ __launch_bounds__(256) void attn_kernel(
    const _Float16* __restrict__ Qb, const _Float16* __restrict__ Kb,
    const _Float16* __restrict__ Vt, float* __restrict__ Out) {
  __shared__ _Float16 Pbuf[8][16 * 32];  // per-wave P staging (1KB each)
  const int lane = threadIdx.x & 31;
  const int w = threadIdx.x >> 5;
  const int g = lane >> 4, ln = lane & 15;
  const int tile = blockIdx.x * 8 + w;  // 0..1023
  const int b = tile >> 7;              // 128 tiles per batch
  const int q0 = (tile & 127) * 16;

  const _Float16* Qrow = Qb + ((size_t)b * S_LEN + q0 + ln) * 64;
  const _Float16* Kbase = Kb + (size_t)b * S_LEN * 64;
  const _Float16* Vbase = Vt + (size_t)b * 64 * S_LEN;
  _Float16* pb = &Pbuf[w][0];

  // Q A-fragments for d_k slices [0,32) and [32,64)
  v16h qf[2];
#pragma unroll
  for (int ks = 0; ks < 2; ++ks)
    qf[ks] = cat16(*(const v8h*)(Qrow + ks * 32 + 16 * g),
                   *(const v8h*)(Qrow + ks * 32 + 16 * g + 8));

  v8f o[4];
  float m[8], l[8];
#pragma unroll
  for (int t = 0; t < 4; ++t)
#pragma unroll
    for (int r = 0; r < 8; ++r) o[t][r] = 0.0f;
#pragma unroll
  for (int r = 0; r < 8; ++r) { m[r] = -1e30f; l[r] = 0.0f; }

  const int nblk = (q0 + 47) >> 5;  // keys 0 .. q0+15 in blocks of 32
  for (int jb = 0; jb < nblk; ++jb) {
    const int j0 = jb * 32;
    v8f sf[2];
#pragma unroll
    for (int f = 0; f < 2; ++f)
#pragma unroll
      for (int r = 0; r < 8; ++r) sf[f][r] = 0.0f;

    // S = Q * K^T : B-frag lane<16: K-row (j0+16nt+ln) dk[ks*32 .. +15]
#pragma unroll
    for (int nt = 0; nt < 2; ++nt)
#pragma unroll
      for (int ks = 0; ks < 2; ++ks) {
        v16h kf = *(const v16h*)(Kbase + (size_t)(j0 + 16 * nt + ln) * 64 +
                                 ks * 32 + 16 * g);
        sf[nt] = wmma16(qf[ks], kf, sf[nt]);
      }

    // Causal mask (elementwise in C layout)
    if (j0 + 31 > q0) {
#pragma unroll
      for (int f = 0; f < 2; ++f)
#pragma unroll
        for (int r = 0; r < 8; ++r) {
          int col = j0 + 16 * f + ln;
          int row = q0 + r + 8 * g;
          if (col > row) sf[f][r] = -1e30f;
        }
    }

    // Online softmax: row reductions across the 16-lane N-group
    float sc[8];
#pragma unroll
    for (int r = 0; r < 8; ++r) {
      float x = fmaxf(sf[0][r], sf[1][r]);
      x = fmaxf(x, __shfl_xor(x, 1, 32));
      x = fmaxf(x, __shfl_xor(x, 2, 32));
      x = fmaxf(x, __shfl_xor(x, 4, 32));
      x = fmaxf(x, __shfl_xor(x, 8, 32));
      float mn = fmaxf(m[r], x);
      sc[r] = __expf(m[r] - mn);
      m[r] = mn;
    }
#pragma unroll
    for (int f = 0; f < 2; ++f)
#pragma unroll
      for (int r = 0; r < 8; ++r) sf[f][r] = __expf(sf[f][r] - m[r]);
#pragma unroll
    for (int r = 0; r < 8; ++r) {
      float x = sf[0][r] + sf[1][r];
      x += __shfl_xor(x, 1, 32);
      x += __shfl_xor(x, 2, 32);
      x += __shfl_xor(x, 4, 32);
      x += __shfl_xor(x, 8, 32);
      l[r] = l[r] * sc[r] + x;
    }
#pragma unroll
    for (int t = 0; t < 4; ++t)
#pragma unroll
      for (int r = 0; r < 8; ++r) o[t][r] *= sc[r];

    // P: C layout -> LDS -> A layout (f16)
#pragma unroll
    for (int r = 0; r < 8; ++r) {
      int row = r + 8 * g;
      pb[row * 32 + ln] = (_Float16)sf[0][r];
      pb[row * 32 + 16 + ln] = (_Float16)sf[1][r];
    }
    __builtin_amdgcn_wave_barrier();
    asm volatile("s_wait_dscnt 0" ::: "memory");
    __builtin_amdgcn_wave_barrier();
    v16h pf = cat16(*(const v8h*)(pb + ln * 32 + 16 * g),
                    *(const v8h*)(pb + ln * 32 + 16 * g + 8));
    __builtin_amdgcn_wave_barrier();

    // O += P(16x32) * V(32x64): B-frag lane<16: Vt row (16t+ln), keys j0..j0+15
#pragma unroll
    for (int t = 0; t < 4; ++t) {
      v16h vf = *(const v16h*)(Vbase + (size_t)(16 * t + ln) * S_LEN + j0 + 16 * g);
      o[t] = wmma16(pf, vf, o[t]);
    }
    asm volatile("s_wait_dscnt 0" ::: "memory");  // P reads done before reuse
  }

  // Epilogue: O / l  (row alignment of l matches C layout)
  float* orow = Out + ((size_t)b * S_LEN + q0) * 64;
#pragma unroll
  for (int r = 0; r < 8; ++r) {
    float inv = 1.0f / l[r];
    int M = r + 8 * g;
#pragma unroll
    for (int t = 0; t < 4; ++t) orow[(size_t)M * 64 + 16 * t + ln] = o[t][r] * inv;
  }
}

// ---------------------------------------------------------------------------
extern "C" void kernel_launch(void* const* d_in, const int* in_sizes, int n_in,
                              void* d_out, int out_size, void* d_ws,
                              size_t ws_size, hipStream_t stream) {
  const float* X = (const float*)d_in[0];
  const float* Wq = (const float*)d_in[1];
  const float* Wk = (const float*)d_in[2];
  const float* Wv = (const float*)d_in[3];
  float* Out = (float*)d_out;

  char* ws = (char*)d_ws;
  _Float16* WT = (_Float16*)(ws);                                   // 384 KB
  _Float16* Qb = (_Float16*)(ws + (512u << 10));                    // 2 MB
  _Float16* Kb = (_Float16*)(ws + (512u << 10) + (2u << 20));       // 2 MB
  _Float16* Vt = (_Float16*)(ws + (512u << 10) + (4u << 20));       // 2 MB

  wt_kernel<<<768, 256, 0, stream>>>(Wq, Wk, Wv, WT);
  qkv_kernel<<<128, 256, 0, stream>>>(X, WT, Qb, Kb, Vt);
  attn_kernel<<<128, 256, 0, stream>>>(Qb, Kb, Vt, Out);
}
